// StructuralEncoderGAT_76879914598743
// MI455X (gfx1250) — matrix operand order repaired and checked by reading, not compiled, not run on hardware
//
#include <hip/hip_runtime.h>
#include <hip/hip_bf16.h>

typedef __attribute__((ext_vector_type(16))) _Float16 v16h;
typedef __attribute__((ext_vector_type(8)))  _Float16 v8h;
typedef __attribute__((ext_vector_type(8)))  float    v8f;

#define NEG_SLOPE 0.2f
#define LN_EPS 1e-5f

// ---------- helpers ----------
__device__ __forceinline__ unsigned enc_f32(float f) {
    unsigned u = __float_as_uint(f);
    return (u & 0x80000000u) ? ~u : (u | 0x80000000u);   // order-preserving key
}
__device__ __forceinline__ float dec_f32(unsigned u) {
    return __uint_as_float((u & 0x80000000u) ? (u & 0x7FFFFFFFu) : ~u);
}
__device__ __forceinline__ float lrelu(float e) { return e >= 0.f ? e : NEG_SLOPE * e; }

union HF { v16h v; _Float16 e[16]; };
union CF { v8f  v; float    e[8];  };

// ---------- GEMM1: h1[nPad,128] = f16(x[N,32]) @ f16(W1[32,128]) via WMMA ----------
// h1 is padded to a multiple of 16 rows -> no store predication needed.
__global__ void __launch_bounds__(256)
k_gemm1(const float* __restrict__ x, const float* __restrict__ W,
        float* __restrict__ h1, int n) {
    // Stage W1 transposed as f16 in LDS: ldsW[c*32 + k] = W[k*128 + c]
    __shared__ _Float16 ldsW[32 * 128];
    for (int i = threadIdx.x; i < 32 * 128; i += blockDim.x) {
        int k = i >> 7, c = i & 127;
        ldsW[c * 32 + k] = (_Float16)W[i];
    }
    __syncthreads();

    const int lane = threadIdx.x & 31;
    const int wave = threadIdx.x >> 5;
    const int row0 = (blockIdx.x * (blockDim.x >> 5) + wave) * 16;
    if (row0 < n) {                               // wave-uniform guard (no early return)
        const int m  = lane & 15;
        const int hi = lane >> 4;                 // K-half select
        int row = row0 + m; if (row >= n) row = n - 1;   // clamp loads only

        // A fragment = two contiguous 8-float chunks per lane (b128 loads)
        const float* xr = x + (long long)row * 32;
        float4 c0 = *(const float4*)(xr + hi * 8);
        float4 c1 = *(const float4*)(xr + hi * 8 + 4);
        float4 c2 = *(const float4*)(xr + 16 + hi * 8);
        float4 c3 = *(const float4*)(xr + 16 + hi * 8 + 4);
        HF A;
        A.e[0]=(_Float16)c0.x; A.e[1]=(_Float16)c0.y; A.e[2]=(_Float16)c0.z; A.e[3]=(_Float16)c0.w;
        A.e[4]=(_Float16)c1.x; A.e[5]=(_Float16)c1.y; A.e[6]=(_Float16)c1.z; A.e[7]=(_Float16)c1.w;
        A.e[8]=(_Float16)c2.x; A.e[9]=(_Float16)c2.y; A.e[10]=(_Float16)c2.z; A.e[11]=(_Float16)c2.w;
        A.e[12]=(_Float16)c3.x; A.e[13]=(_Float16)c3.y; A.e[14]=(_Float16)c3.z; A.e[15]=(_Float16)c3.w;

        const int kb = hi * 8;
        #pragma unroll
        for (int t = 0; t < 8; ++t) {             // 8 column tiles of 16 (128 out)
            const int col = t * 16 + m;
            HF B;                                  // two 8-half chunks (k=8h.., 16+8h..) from LDS
            *(v8h*)&B.e[0] = *(const v8h*)(ldsW + col * 32 + kb);
            *(v8h*)&B.e[8] = *(const v8h*)(ldsW + col * 32 + kb + 16);
            CF C; C.v = (v8f){0.f,0.f,0.f,0.f,0.f,0.f,0.f,0.f};
            C.v = __builtin_amdgcn_wmma_f32_16x16x32_f16(false, A.v, false, B.v,
                                                         (short)0, C.v, false, false);
            float* hp = h1 + (long long)(row0 + hi * 8) * 128 + col;
            #pragma unroll
            for (int v = 0; v < 8; ++v) hp[(long long)v * 128] = C.e[v];  // padded: no bounds
        }
    }
}

// ---------- GEMM2: h2[nPad,64] = f16(hin[nPad,128]) @ f16(W2[128,64]) ----------
__global__ void __launch_bounds__(256)
k_gemm2(const float* __restrict__ hin, const float* __restrict__ W,
        float* __restrict__ h2, int n) {
    // Stage W2 transposed as f16 in LDS: ldsW[c*128 + k] = W[k*64 + c]
    __shared__ _Float16 ldsW[128 * 64];
    for (int i = threadIdx.x; i < 128 * 64; i += blockDim.x) {
        int k = i >> 6, c = i & 63;
        ldsW[c * 128 + k] = (_Float16)W[i];
    }
    __syncthreads();

    const int lane = threadIdx.x & 31;
    const int wave = threadIdx.x >> 5;
    const int row0 = (blockIdx.x * (blockDim.x >> 5) + wave) * 16;
    if (row0 < n) {
        const int m  = lane & 15;
        const int hi = lane >> 4;
        int row = row0 + m; if (row >= n) row = n - 1;

        HF A[4];
        const float* hr = hin + (long long)row * 128;
        #pragma unroll
        for (int kk = 0; kk < 4; ++kk) {
            float4 c0 = *(const float4*)(hr + kk * 32 + hi * 8);
            float4 c1 = *(const float4*)(hr + kk * 32 + hi * 8 + 4);
            float4 c2 = *(const float4*)(hr + kk * 32 + 16 + hi * 8);
            float4 c3 = *(const float4*)(hr + kk * 32 + 16 + hi * 8 + 4);
            A[kk].e[0]=(_Float16)c0.x; A[kk].e[1]=(_Float16)c0.y; A[kk].e[2]=(_Float16)c0.z; A[kk].e[3]=(_Float16)c0.w;
            A[kk].e[4]=(_Float16)c1.x; A[kk].e[5]=(_Float16)c1.y; A[kk].e[6]=(_Float16)c1.z; A[kk].e[7]=(_Float16)c1.w;
            A[kk].e[8]=(_Float16)c2.x; A[kk].e[9]=(_Float16)c2.y; A[kk].e[10]=(_Float16)c2.z; A[kk].e[11]=(_Float16)c2.w;
            A[kk].e[12]=(_Float16)c3.x; A[kk].e[13]=(_Float16)c3.y; A[kk].e[14]=(_Float16)c3.z; A[kk].e[15]=(_Float16)c3.w;
        }
        const int kb = hi * 8;
        #pragma unroll
        for (int t = 0; t < 4; ++t) {             // 4 column tiles (64 out)
            const int col = t * 16 + m;
            CF C; C.v = (v8f){0.f,0.f,0.f,0.f,0.f,0.f,0.f,0.f};
            #pragma unroll
            for (int kk = 0; kk < 4; ++kk) {      // K = 128 = 4 x 32
                HF B;
                *(v8h*)&B.e[0] = *(const v8h*)(ldsW + col * 128 + kk * 32 + kb);
                *(v8h*)&B.e[8] = *(const v8h*)(ldsW + col * 128 + kk * 32 + kb + 16);
                C.v = __builtin_amdgcn_wmma_f32_16x16x32_f16(false, A[kk].v, false, B.v,
                                                             (short)0, C.v, false, false);
            }
            float* hp = h2 + (long long)(row0 + hi * 8) * 64 + col;
            #pragma unroll
            for (int v = 0; v < 8; ++v) hp[(long long)v * 64] = C.e[v];   // padded: no bounds
        }
    }
}

// ---------- init kernels (float4 stores) ----------
__global__ void k_init1(float4* __restrict__ acc, const float4* __restrict__ bias,
                        unsigned* __restrict__ emax, float* __restrict__ esum, int n) {
    long long idx = (long long)blockIdx.x * blockDim.x + threadIdx.x;
    if (idx >= (long long)n * 32) return;
    int q = (int)(idx & 31);
    acc[idx] = bias[q];
    if (q == 0) {
        long long node = idx >> 5;
        emax[node * 2]     = enc_f32(-1e30f);
        emax[node * 2 + 1] = enc_f32(-1e30f);
        esum[node * 2] = 0.f; esum[node * 2 + 1] = 0.f;
    }
}
__global__ void k_init2(float4* __restrict__ out, const float4* __restrict__ bias,
                        unsigned* __restrict__ emax, float* __restrict__ esum, int n) {
    long long idx = (long long)blockIdx.x * blockDim.x + threadIdx.x;
    if (idx >= (long long)n * 16) return;
    int q = (int)(idx & 15);
    out[idx] = bias[q];
    if (q == 0) { long long node = idx >> 4; emax[node] = enc_f32(-1e30f); esum[node] = 0.f; }
}

// ---------- attention coefficients (wave per node, lane = 4 channels) ----------
__global__ void k_att1(const float* __restrict__ h1, const float* __restrict__ aw_s,
                       const float* __restrict__ aw_d, float* __restrict__ a_s,
                       float* __restrict__ a_d, int n) {
    int wid  = (int)(((long long)blockIdx.x * blockDim.x + threadIdx.x) >> 5);
    int lane = threadIdx.x & 31;
    if (wid >= n) return;
    float4 v  = *(const float4*)(h1 + (long long)wid * 128 + lane * 4);
    float4 ws = *(const float4*)(aw_s + lane * 4);      // att [2,64] flat == channel idx
    float4 wd = *(const float4*)(aw_d + lane * 4);
    float s = v.x*ws.x + v.y*ws.y + v.z*ws.z + v.w*ws.w;
    float d = v.x*wd.x + v.y*wd.y + v.z*wd.z + v.w*wd.w;
    #pragma unroll
    for (int off = 8; off; off >>= 1) {                 // reduce within 16-lane head group
        s += __shfl_xor(s, off, 32); d += __shfl_xor(d, off, 32);
    }
    if ((lane & 15) == 0) {
        int h = lane >> 4;                              // lanes 0-15: head0, 16-31: head1
        a_s[wid * 2 + h] = s; a_d[wid * 2 + h] = d;
    }
}
__global__ void k_att2(const float* __restrict__ h2, const float* __restrict__ aw_s,
                       const float* __restrict__ aw_d, float* __restrict__ a_s,
                       float* __restrict__ a_d, int n) {
    int wid  = (int)(((long long)blockIdx.x * blockDim.x + threadIdx.x) >> 5);
    int lane = threadIdx.x & 31;
    if (wid >= n) return;
    float2 v  = *(const float2*)(h2 + (long long)wid * 64 + lane * 2);
    float2 ws = *(const float2*)(aw_s + lane * 2);
    float2 wd = *(const float2*)(aw_d + lane * 2);
    float s = v.x*ws.x + v.y*ws.y;
    float d = v.x*wd.x + v.y*wd.y;
    #pragma unroll
    for (int off = 16; off; off >>= 1) {
        s += __shfl_xor(s, off, 32); d += __shfl_xor(d, off, 32);
    }
    if (lane == 0) { a_s[wid] = s; a_d[wid] = d; }
}

// ---------- softmax pass 1: segment max (thread per edge, both heads) ----------
__global__ void k_edge_max1(const int* __restrict__ ei, int E, int n,
                            const float* __restrict__ a_s, const float* __restrict__ a_d,
                            unsigned* __restrict__ emax) {
    long long i = (long long)blockIdx.x * blockDim.x + threadIdx.x;
    if (i >= (long long)E + n) return;
    int s, d;
    if (i < (long long)E) { s = ei[i]; d = ei[(long long)E + i]; } else { s = d = (int)(i - E); }
    float2 as = *(const float2*)(a_s + (long long)s * 2);
    float2 ad = *(const float2*)(a_d + (long long)d * 2);
    atomicMax(&emax[(long long)d * 2],     enc_f32(lrelu(as.x + ad.x)));
    atomicMax(&emax[(long long)d * 2 + 1], enc_f32(lrelu(as.y + ad.y)));
}
__global__ void k_edge_max2(const int* __restrict__ ei, int E, int n,
                            const float* __restrict__ a_s, const float* __restrict__ a_d,
                            unsigned* __restrict__ emax) {
    long long i = (long long)blockIdx.x * blockDim.x + threadIdx.x;
    if (i >= (long long)E + n) return;
    int s, d;
    if (i < (long long)E) { s = ei[i]; d = ei[(long long)E + i]; } else { s = d = (int)(i - E); }
    atomicMax(&emax[d], enc_f32(lrelu(a_s[s] + a_d[d])));
}

// ---------- softmax pass 2: sum of exp (thread per edge, both heads) ----------
__global__ void k_edge_sum1(const int* __restrict__ ei, int E, int n,
                            const float* __restrict__ a_s, const float* __restrict__ a_d,
                            const unsigned* __restrict__ emax, float* __restrict__ esum) {
    long long i = (long long)blockIdx.x * blockDim.x + threadIdx.x;
    if (i >= (long long)E + n) return;
    int s, d;
    if (i < (long long)E) { s = ei[i]; d = ei[(long long)E + i]; } else { s = d = (int)(i - E); }
    float2 as = *(const float2*)(a_s + (long long)s * 2);
    float2 ad = *(const float2*)(a_d + (long long)d * 2);
    uint2  mx = *(const uint2*)(emax + (long long)d * 2);
    atomicAdd(&esum[(long long)d * 2],     __expf(lrelu(as.x + ad.x) - dec_f32(mx.x)));
    atomicAdd(&esum[(long long)d * 2 + 1], __expf(lrelu(as.y + ad.y) - dec_f32(mx.y)));
}
__global__ void k_edge_sum2(const int* __restrict__ ei, int E, int n,
                            const float* __restrict__ a_s, const float* __restrict__ a_d,
                            const unsigned* __restrict__ emax, float* __restrict__ esum) {
    long long i = (long long)blockIdx.x * blockDim.x + threadIdx.x;
    if (i >= (long long)E + n) return;
    int s, d;
    if (i < (long long)E) { s = ei[i]; d = ei[(long long)E + i]; } else { s = d = (int)(i - E); }
    atomicAdd(&esum[d], __expf(lrelu(a_s[s] + a_d[d]) - dec_f32(emax[d])));
}

// ---------- softmax pass 3: weighted scatter (wave per edge, lane = 4 channels) ----------
// Edge id is wave-uniform: force it into an SGPR so edge indices and softmax
// coefficients load through the scalar path (SMEM), keeping VMEM for the row.
__global__ void k_edge_msg1(const int* __restrict__ ei, int E, int n,
                            const float* __restrict__ a_s, const float* __restrict__ a_d,
                            const unsigned* __restrict__ emax, const float* __restrict__ esum,
                            const float* __restrict__ h1, float* __restrict__ acc) {
    const int wi = __builtin_amdgcn_readfirstlane(
        (int)(((long long)blockIdx.x * blockDim.x + threadIdx.x) >> 5));
    const int lane = threadIdx.x & 31;
    if (wi >= E + n) return;
    int s, d;
    if (wi < E) { s = ei[wi]; d = ei[(long long)E + wi]; }   // uniform -> s_load
    else        { s = d = wi - E; }
    const float* srow = h1 + (long long)s * 128 + lane * 4;
    __builtin_prefetch(srow, 0, 0);                     // global_prefetch for the gather row
    float as0 = a_s[(long long)s * 2], as1 = a_s[(long long)s * 2 + 1];
    float ad0 = a_d[(long long)d * 2], ad1 = a_d[(long long)d * 2 + 1];
    unsigned m0 = emax[(long long)d * 2], m1 = emax[(long long)d * 2 + 1];
    float sm0 = esum[(long long)d * 2], sm1 = esum[(long long)d * 2 + 1];
    const int h = lane >> 4;                            // channels lane*4.. are all head h
    float e  = lrelu((h ? as1 : as0) + (h ? ad1 : ad0));
    float al = __expf(e - dec_f32(h ? m1 : m0)) / ((h ? sm1 : sm0) + 1e-16f);
    float4 hv = *(const float4*)srow;
    float* dp = acc + (long long)d * 128 + lane * 4;
    atomicAdd(dp + 0, hv.x * al);
    atomicAdd(dp + 1, hv.y * al);
    atomicAdd(dp + 2, hv.z * al);
    atomicAdd(dp + 3, hv.w * al);
}
__global__ void k_edge_msg2(const int* __restrict__ ei, int E, int n,
                            const float* __restrict__ a_s, const float* __restrict__ a_d,
                            const unsigned* __restrict__ emax, const float* __restrict__ esum,
                            const float* __restrict__ h2, float* __restrict__ out) {
    const int wi = __builtin_amdgcn_readfirstlane(
        (int)(((long long)blockIdx.x * blockDim.x + threadIdx.x) >> 5));
    const int lane = threadIdx.x & 31;
    if (wi >= E + n) return;
    int s, d;
    if (wi < E) { s = ei[wi]; d = ei[(long long)E + wi]; }
    else        { s = d = wi - E; }
    const float* srow = h2 + (long long)s * 64 + lane * 2;
    __builtin_prefetch(srow, 0, 0);
    float e  = lrelu(a_s[s] + a_d[d]);
    float al = __expf(e - dec_f32(emax[d])) / (esum[d] + 1e-16f);
    float2 hv = *(const float2*)srow;
    float* dp = out + (long long)d * 64 + lane * 2;
    atomicAdd(dp + 0, hv.x * al);
    atomicAdd(dp + 1, hv.y * al);
}

// ---------- LayerNorm(128) + ReLU, in place (wave per node, float4) ----------
__global__ void k_ln_relu(float* __restrict__ acc, const float* __restrict__ gamma,
                          const float* __restrict__ beta, int n) {
    int wid  = (int)(((long long)blockIdx.x * blockDim.x + threadIdx.x) >> 5);
    int lane = threadIdx.x & 31;
    if (wid >= n) return;
    float* row = acc + (long long)wid * 128;
    float4 v = *(const float4*)(row + lane * 4);
    float s  = v.x + v.y + v.z + v.w;
    float sq = v.x*v.x + v.y*v.y + v.z*v.z + v.w*v.w;
    #pragma unroll
    for (int off = 16; off; off >>= 1) {
        s += __shfl_xor(s, off, 32); sq += __shfl_xor(sq, off, 32);
    }
    float mu  = s * (1.f / 128.f);
    float var = sq * (1.f / 128.f) - mu * mu;
    float rs  = rsqrtf(var + LN_EPS);
    float4 g = *(const float4*)(gamma + lane * 4);
    float4 b = *(const float4*)(beta + lane * 4);
    float4 o;
    o.x = (v.x - mu) * rs * g.x + b.x; o.x = o.x > 0.f ? o.x : 0.f;
    o.y = (v.y - mu) * rs * g.y + b.y; o.y = o.y > 0.f ? o.y : 0.f;
    o.z = (v.z - mu) * rs * g.z + b.z; o.z = o.z > 0.f ? o.z : 0.f;
    o.w = (v.w - mu) * rs * g.w + b.w; o.w = o.w > 0.f ? o.w : 0.f;
    *(float4*)(row + lane * 4) = o;
}

static inline unsigned cdiv(long long a, long long b) { return (unsigned)((a + b - 1) / b); }

extern "C" void kernel_launch(void* const* d_in, const int* in_sizes, int n_in,
                              void* d_out, int out_size, void* d_ws, size_t ws_size,
                              hipStream_t stream) {
    const float* x     = (const float*)d_in[0];
    const int*   ei    = (const int*)  d_in[1];
    const float* W1    = (const float*)d_in[2];
    const float* as1w  = (const float*)d_in[3];
    const float* ad1w  = (const float*)d_in[4];
    const float* b1    = (const float*)d_in[5];
    const float* gamma = (const float*)d_in[6];
    const float* beta  = (const float*)d_in[7];
    const float* W2    = (const float*)d_in[8];
    const float* as2w  = (const float*)d_in[9];
    const float* ad2w  = (const float*)d_in[10];
    const float* b2    = (const float*)d_in[11];
    float* out = (float*)d_out;

    const int n = in_sizes[0] / 32;
    const int E = in_sizes[1] / 2;
    const int nPad = (n + 15) & ~15;                 // padded rows for GEMM outputs
    const long long ET = (long long)E + n;

    float* ws = (float*)d_ws;
    float* h1   = ws;  ws += (size_t)nPad * 128;     // padded (gemm1 stores full tiles)
    float* acc1 = ws;  ws += (size_t)nPad * 128;     // padded (gemm2 loads clamp rows)
    float* h2   = ws;  ws += (size_t)nPad * 64;      // padded (gemm2 stores full tiles)
    float* a_s1 = ws;  ws += (size_t)n * 2;
    float* a_d1 = ws;  ws += (size_t)n * 2;
    unsigned* emax1 = (unsigned*)ws; ws += (size_t)n * 2;
    float* esum1 = ws; ws += (size_t)n * 2;
    float* a_s2 = ws;  ws += n;
    float* a_d2 = ws;  ws += n;
    unsigned* emax2 = (unsigned*)ws; ws += n;
    float* esum2 = ws; ws += n;

    const int B = 256;
    const unsigned gTiles = cdiv(cdiv((long long)n, 16), 8);    // 8 waves/block
    const unsigned gEdge  = cdiv(ET, B);                        // thread per edge
    const unsigned gEdgeW = cdiv(ET * 32, B);                   // wave per edge
    const unsigned gNodeW = cdiv((long long)n * 32, B);         // wave per node

    // ---- layer 1 ----
    k_init1<<<cdiv((long long)n * 32, B), B, 0, stream>>>((float4*)acc1, (const float4*)b1,
                                                          emax1, esum1, n);
    k_gemm1<<<gTiles, B, 0, stream>>>(x, W1, h1, n);
    k_att1 <<<gNodeW, B, 0, stream>>>(h1, as1w, ad1w, a_s1, a_d1, n);
    k_edge_max1<<<gEdge, B, 0, stream>>>(ei, E, n, a_s1, a_d1, emax1);
    k_edge_sum1<<<gEdge, B, 0, stream>>>(ei, E, n, a_s1, a_d1, emax1, esum1);
    k_edge_msg1<<<gEdgeW, B, 0, stream>>>(ei, E, n, a_s1, a_d1, emax1, esum1, h1, acc1);
    k_ln_relu<<<gNodeW, B, 0, stream>>>(acc1, gamma, beta, n);

    // ---- layer 2 ----
    k_init2<<<cdiv((long long)n * 16, B), B, 0, stream>>>((float4*)out, (const float4*)b2,
                                                          emax2, esum2, n);
    k_gemm2<<<gTiles, B, 0, stream>>>(acc1, W2, h2, n);
    k_att2 <<<gNodeW, B, 0, stream>>>(h2, as2w, ad2w, a_s2, a_d2, n);
    k_edge_max2<<<gEdge, B, 0, stream>>>(ei, E, n, a_s2, a_d2, emax2);
    k_edge_sum2<<<gEdge, B, 0, stream>>>(ei, E, n, a_s2, a_d2, emax2, esum2);
    k_edge_msg2<<<gEdgeW, B, 0, stream>>>(ei, E, n, a_s2, a_d2, emax2, esum2, h2, out);
}
